// RefillModelOld_29343216566612
// MI455X (gfx1250) — matrix operand order repaired and checked by reading, not compiled, not run on hardware
//
#include <hip/hip_runtime.h>
#include <stdint.h>

#define B_ 8
#define L_ 256
#define LY_ 64
#define V_ 32000
#define E_ 128
#define M_ 16
#define N_ 80  // M_ + LY_

typedef __attribute__((ext_vector_type(16))) __bf16 v16bf;
typedef __attribute__((ext_vector_type(8)))  float  v8f;

union BFrag { v16bf v; uint4 q[2]; };

// ---------------------------------------------------------------- helpers ----
__device__ __forceinline__ uint16_t f2bf(float f) {
  union { float f; uint32_t u; } c; c.f = f;
  uint32_t u = c.u;
  uint32_t r = (u + 0x7FFFu + ((u >> 16) & 1u)) >> 16;  // RNE
  return (uint16_t)r;
}

// 1/(1e-5 + std(p[0..127])) with ddof=1 (matches jnp.std default)
__device__ __forceinline__ float row_inv_std(const float* p) {
  float s1 = 0.f, s2 = 0.f;
  for (int k = 0; k < E_; ++k) { float v = p[k]; s1 += v; s2 += v * v; }
  float var = (s2 - s1 * s1 * (1.f / E_)) * (1.f / (E_ - 1));
  var = fmaxf(var, 0.f);
  return 1.f / (1e-5f + sqrtf(var));
}

// ------------------------------------------------- kernel 1a: embed -> bf16 --
__global__ __launch_bounds__(256) void k_cvt_bf16(const float* __restrict__ src,
                                                  uint16_t* __restrict__ dst) {
  int i = (blockIdx.x * 256 + threadIdx.x) * 4;
  float4 v = *(const float4*)(src + i);
  ushort4 o;
  o.x = f2bf(v.x); o.y = f2bf(v.y); o.z = f2bf(v.z); o.w = f2bf(v.w);
  *(ushort4*)(dst + i) = o;
}

// --------------------------------------- kernel 1b: gather + norm (z_e, y_e) --
__global__ __launch_bounds__(128) void k_gather_norm(
    const float* __restrict__ embed, const int* __restrict__ zidx,
    const int* __restrict__ yidx, float* __restrict__ ze, float* __restrict__ ye) {
  int r = blockIdx.x, t = threadIdx.x;
  int tok; float* dst;
  if (r < B_ * L_) { tok = zidx[r]; dst = ze + (size_t)r * E_; }
  else { int r2 = r - B_ * L_; tok = yidx[r2]; dst = ye + (size_t)r2 * E_; }
  float v = embed[(size_t)tok * E_ + t];
  float s1 = v, s2 = v * v;
  #pragma unroll
  for (int off = 16; off > 0; off >>= 1) {
    s1 += __shfl_xor(s1, off, 32);
    s2 += __shfl_xor(s2, off, 32);
  }
  __shared__ float w1[4], w2[4];
  __shared__ float sinv;
  if ((t & 31) == 0) { w1[t >> 5] = s1; w2[t >> 5] = s2; }
  __syncthreads();
  if (t == 0) {
    float a = 0.f, q = 0.f;
    for (int i = 0; i < 4; ++i) { a += w1[i]; q += w2[i]; }
    float var = (q - a * a * (1.f / E_)) * (1.f / (E_ - 1));
    sinv = 1.f / (1e-5f + sqrtf(fmaxf(var, 0.f)));
  }
  __syncthreads();
  dst[t] = v * sinv;
}

// -------------------------- kernel 2: sequential scan, 1 WG (WGP) per batch --
// All persistent weights live in LDS (~245 KB of the 320 KB CDNA5 WGP LDS).
__global__ __launch_bounds__(256) void k_scan(
    const float* __restrict__ ze, const float* __restrict__ ye,
    const float* __restrict__ init_state_w,
    const float* __restrict__ updater_w, const float* __restrict__ updater_b,
    const float* __restrict__ updater_v_w, const float* __restrict__ updater_v_b,
    const float* __restrict__ query_init_w,
    const float* __restrict__ query_k_w, const float* __restrict__ query_k_b,
    const float* __restrict__ query_v_w, const float* __restrict__ query_v_b,
    float* __restrict__ fs32, uint16_t* __restrict__ fsb) {
  extern __shared__ float sm[];
  float* Wu     = sm;                 // 16384
  float* Wv     = Wu + E_ * E_;       // 16384
  float* Wq     = Wv + E_ * E_;       // 16384
  float* xas    = Wq + E_ * E_;       // 80*128: rows 0..15 = xs, rows 16..79 = y_e
  float* zt     = xas + N_ * E_;      // 128
  float* u      = zt + E_;
  float* zv     = u + E_;
  float* qk     = zv + E_;
  float* bu     = qk + E_;
  float* bv     = bu + E_;
  float* bq     = bv + E_;
  float* xv     = bq + E_;
  float* xq     = xv + E_;
  float* red1   = xq + E_;            // 256
  float* red2   = red1 + 256;         // 256
  float* rowinv = red2 + 256;         // 16
  float* att    = rowinv + M_;        // 16
  float* qlog   = att + M_;           // 80
  float* qex    = qlog + N_;          // 80
  float* scal   = qex + N_;           // 8

  const int t = threadIdx.x;
  const int b = blockIdx.x;

  // ---- init: stage weights / constants into LDS ----
  for (int i = t; i < E_ * E_; i += 256) {
    Wu[i] = updater_w[i];
    Wv[i] = updater_v_w[i];
    Wq[i] = query_v_w[i];
  }
  for (int i = t; i < LY_ * E_; i += 256)
    xas[M_ * E_ + i] = ye[(size_t)b * LY_ * E_ + i];
  if (t < E_) { bu[t] = updater_b[t]; bv[t] = updater_v_b[t]; bq[t] = query_v_b[t]; }
  for (int i = t; i < M_ * E_; i += 256) {            // xs0 raw = init_state_w.T
    int m = i >> 7, e = i & (E_ - 1);
    xas[i] = init_state_w[e * M_ + m];
  }
  if (t < E_) xv[t] = query_init_w[t];                // raw query init
  __syncthreads();
  if (t < M_) rowinv[t] = row_inv_std(xas + t * E_);
  if (t == 31) scal[0] = row_inv_std(xv);
  __syncthreads();
  for (int i = t; i < M_ * E_; i += 256) xas[i] *= rowinv[i >> 7];
  if (t < E_) xv[t] *= scal[0];                       // nq = norm(query_init)
  __syncthreads();
  if (t < E_) {                                       // qk = nq @ Wk^T + bk
    float acc = query_k_b[t];
    const float* wr = query_k_w + t * E_;
    for (int k = 0; k < E_; ++k) acc += xv[k] * wr[k];
    qk[t] = acc;
  }
  __syncthreads();

  // ---- 256 sequential steps ----
  for (int step = 0; step < L_; ++step) {
    if (t < E_) zt[t] = ze[((size_t)b * L_ + step) * E_ + t];
    __syncthreads();
    // u = zt@Wu^T+bu (threads 0..127), zv = zt@Wv^T+bv (threads 128..255);
    // rotation (k+e)&127 -> conflict-free LDS banks
    {
      int e = t & (E_ - 1);
      const float* W = (t < E_) ? Wu : Wv;
      float acc = (t < E_) ? bu[e] : bv[e];
      const float* wr = W + e * E_;
      for (int k = 0; k < E_; ++k) { int kk = (k + e) & (E_ - 1); acc += zt[kk] * wr[kk]; }
      if (t < E_) u[e] = acc; else zv[e] = acc;
    }
    __syncthreads();
    // att logits over M=16 state rows
    {
      int m = t >> 4, j = t & 15;
      const float* xr = xas + m * E_ + j * 8;
      const float* ur = u + j * 8;
      float p = 0.f;
      #pragma unroll
      for (int k = 0; k < 8; ++k) p += ur[k] * xr[k];
      red1[t] = p;
    }
    __syncthreads();
    if (t < M_) { float s = 0.f; for (int j = 0; j < 16; ++j) s += red1[t * 16 + j]; red2[t] = s; }
    __syncthreads();
    if (t == 0) {                                     // softmax over 16
      float mx = -1e30f;
      for (int m = 0; m < M_; ++m) mx = fmaxf(mx, red2[m]);
      float s = 0.f;
      for (int m = 0; m < M_; ++m) { float e = __expf(red2[m] - mx); att[m] = e; s += e; }
      float inv = 1.f / s;
      for (int m = 0; m < M_; ++m) att[m] *= inv;
    }
    __syncthreads();
    // xs += att*zv, accumulate row stats
    {
      int m = t >> 4, j = t & 15;
      float a = att[m];
      float* xr = xas + m * E_ + j * 8;
      const float* zr = zv + j * 8;
      float s1 = 0.f, s2 = 0.f;
      #pragma unroll
      for (int k = 0; k < 8; ++k) {
        float v = xr[k] + a * zr[k];
        xr[k] = v; s1 += v; s2 += v * v;
      }
      red1[t] = s1; red2[t] = s2;
    }
    __syncthreads();
    if (t < M_) {
      float s1 = 0.f, s2 = 0.f;
      for (int j = 0; j < 16; ++j) { s1 += red1[t * 16 + j]; s2 += red2[t * 16 + j]; }
      float var = (s2 - s1 * s1 * (1.f / E_)) * (1.f / (E_ - 1));
      rowinv[t] = 1.f / (1e-5f + sqrtf(fmaxf(var, 0.f)));
    }
    __syncthreads();
    { int m = t >> 4, j = t & 15; float inv = rowinv[m];
      float* xr = xas + m * E_ + j * 8;
      #pragma unroll
      for (int k = 0; k < 8; ++k) xr[k] *= inv; }
    __syncthreads();
    // query attention logits over N=80 rows (2 threads/row, bank-rotated)
    if (t < 2 * N_) {
      int n = t >> 1, h = t & 1;
      const float* xr = xas + n * E_ + h * 64;
      const float* qr = qk + h * 64;
      float p = 0.f;
      for (int k = 0; k < 64; ++k) { int kk = (k + 2 * t) & 63; p += qr[kk] * xr[kk]; }
      red1[t] = p;
    }
    __syncthreads();
    if (t < N_) qlog[t] = red1[2 * t] + red1[2 * t + 1];
    __syncthreads();
    if (t == 0) { float mx = -1e30f; for (int n = 0; n < N_; ++n) mx = fmaxf(mx, qlog[n]); scal[1] = mx; }
    __syncthreads();
    if (t < N_) qex[t] = __expf(qlog[t] - scal[1]);
    __syncthreads();
    if (t == 0) { float s = 0.f; for (int n = 0; n < N_; ++n) s += qex[n]; scal[2] = 1.f / s; }
    __syncthreads();
    // xv = softmax(q) @ xas  (80x128 contraction; column access = conflict-free)
    {
      int e = t & (E_ - 1), h = t >> 7;
      float p = 0.f;
      for (int n = h * 40; n < h * 40 + 40; ++n) p += qex[n] * xas[n * E_ + e];
      red1[t] = p;
    }
    __syncthreads();
    if (t < E_) xv[t] = (red1[t] + red1[t + E_]) * scal[2];
    __syncthreads();
    // xq = xv @ Wq^T + bq   (query_v applied AFTER the contraction: same result, 80x fewer matvecs)
    if (t < E_) {
      int e = t;
      float acc = bq[e];
      const float* wr = Wq + e * E_;
      for (int k = 0; k < E_; ++k) { int kk = (k + e) & (E_ - 1); acc += xv[kk] * wr[kk]; }
      xq[e] = acc;
    }
    __syncthreads();
    if (t == 0) scal[3] = row_inv_std(xq);
    __syncthreads();
    if (t < E_) {
      float o = xq[t] * scal[3];
      size_t idx = ((size_t)b * L_ + step) * E_ + t;
      fs32[idx] = o;
      fsb[idx] = f2bf(o);
    }
    __syncthreads();
  }
}

// ------------- kernel 3: WMMA bf16 LM-head GEMM with fused sum-exp (lse) -----
// 16 rows per WG, 8 waves sweep V in 16-col tiles; A frags stay in VGPRs;
// B tiles double-buffered so next tile's 8x b128 loads overlap current WMMAs.
// No online max: fs rows are unit-std (||fs||~=11.3), embed rows ~N(0,1/E)
// (||e|| <~ 1.7), so |logit| <= ||fs||*||e|| < ~20 -> exp/sum safely in f32.
__device__ __forceinline__ void load_btile(const uint16_t* __restrict__ brow,
                                           BFrag* __restrict__ f) {
  #pragma unroll
  for (int c = 0; c < 4; ++c) {
    f[c].q[0] = *(const uint4*)(brow + 32 * c);
    f[c].q[1] = *(const uint4*)(brow + 32 * c + 16);
  }
}

__global__ __launch_bounds__(256) void k_lse(const uint16_t* __restrict__ fsb,
                                             const uint16_t* __restrict__ embb,
                                             float* __restrict__ lse) {
  __shared__ float rs[8][16];
  const int tid = threadIdx.x;
  const int wave = tid >> 5, lane = tid & 31;
  const int mrow = lane & 15, kh = lane >> 4;

  // A fragments (16x32 bf16, K-chunks c=0..3): per CDNA5 layout lane holds
  // K = kh*8 + {0..7} (VGPR0..3) and K = 16 + kh*8 + {0..7} (VGPR4..7).
  const uint16_t* arow = fsb + (size_t)(blockIdx.x * 16 + mrow) * E_;
  BFrag a[4];
  #pragma unroll
  for (int c = 0; c < 4; ++c) {
    a[c].q[0] = *(const uint4*)(arow + 32 * c + kh * 8);
    a[c].q[1] = *(const uint4*)(arow + 32 * c + 16 + kh * 8);
  }
  float lsum[8];
  #pragma unroll
  for (int j = 0; j < 8; ++j) lsum[j] = 0.f;

  const int NIT = (V_ / 16) / 8;           // 250 tiles per wave (even)
  const size_t STRIDE = (size_t)128 * E_;  // +8 tiles = 128 columns
  // B fragment (32x16 bf16): lane holds K = kh*8 + {0..7} and K = 16 + kh*8 + {0..7} of its column.
  const uint16_t* p0 = embb + (size_t)(wave * 16 + mrow) * E_ + kh * 8;

  BFrag b0[4], b1[4];
  load_btile(p0, b0);
  for (int it = 0; it < NIT; it += 2) {
    const uint16_t* p1 = p0 + STRIDE;
    load_btile(p1, b1);                    // tile it+1 (NIT even -> always valid)
    if (it + 4 < NIT) __builtin_prefetch(p1 + STRIDE, 0, 1);
    v8f acc = {};
    #pragma unroll
    for (int c = 0; c < 4; ++c)
      acc = __builtin_amdgcn_wmma_f32_16x16x32_bf16(false, a[c].v, false, b0[c].v,
                                                    (short)0, acc, false, false);
    #pragma unroll
    for (int j = 0; j < 8; ++j) lsum[j] += __expf(acc[j]);
    p0 = p1 + STRIDE;
    if (it + 2 < NIT) load_btile(p0, b0);  // tile it+2 overlaps WMMAs below
    v8f acc2 = {};
    #pragma unroll
    for (int c = 0; c < 4; ++c)
      acc2 = __builtin_amdgcn_wmma_f32_16x16x32_bf16(false, a[c].v, false, b1[c].v,
                                                     (short)0, acc2, false, false);
    #pragma unroll
    for (int j = 0; j < 8; ++j) lsum[j] += __expf(acc2[j]);
  }
  // sum across the 16 lanes sharing each row group (lane owns rows kh*8+j)
  #pragma unroll
  for (int off = 1; off < 16; off <<= 1) {
    #pragma unroll
    for (int j = 0; j < 8; ++j) lsum[j] += __shfl_xor(lsum[j], off, 32);
  }
  if (mrow == 0) {
    #pragma unroll
    for (int j = 0; j < 8; ++j) rs[wave][kh * 8 + j] = lsum[j];
  }
  __syncthreads();
  if (tid < 16) {  // merge across the 8 waves
    float s = 0.f;
    for (int w = 0; w < 8; ++w) s += rs[w][tid];
    lse[blockIdx.x * 16 + tid] = __logf(s);
  }
}

// ------------------- kernel 4: target logit + per-batch mean NLL reduction ---
__global__ __launch_bounds__(256) void k_final(
    const float* __restrict__ fs32, const float* __restrict__ embed,
    const int* __restrict__ x, const float* __restrict__ lse,
    float* __restrict__ out) {
  __shared__ float red[256];
  int b = blockIdx.x, l = threadIdx.x;
  int row = b * L_ + l;
  int tok = x[row];
  const float* fr = fs32 + (size_t)row * E_;
  const float* er = embed + (size_t)tok * E_;
  float d = 0.f;
  for (int k = 0; k < E_; ++k) d += fr[k] * er[k];
  red[l] = lse[row] - d;  // -log p(target)
  __syncthreads();
  for (int s = 128; s > 0; s >>= 1) {
    if (l < s) red[l] += red[l + s];
    __syncthreads();
  }
  if (l == 0) out[b] = red[0] * (1.f / L_);
}

// ----------------------------------------------------------------- launch ----
static constexpr size_t SCAN_LDS_BYTES =
    (size_t)(3 * E_ * E_ + N_ * E_ + 9 * E_ + 2 * 256 + 2 * M_ + 2 * N_ + 8) * sizeof(float);

extern "C" void kernel_launch(void* const* d_in, const int* in_sizes, int n_in,
                              void* d_out, int out_size, void* d_ws, size_t ws_size,
                              hipStream_t stream) {
  (void)in_sizes; (void)n_in; (void)out_size; (void)ws_size;
  const int*   x            = (const int*)  d_in[1];
  const int*   y            = (const int*)  d_in[2];
  const int*   z            = (const int*)  d_in[3];
  const float* embed_w      = (const float*)d_in[4];
  const float* init_state_w = (const float*)d_in[5];
  const float* updater_w    = (const float*)d_in[6];
  const float* updater_b    = (const float*)d_in[7];
  const float* updater_v_w  = (const float*)d_in[8];
  const float* updater_v_b  = (const float*)d_in[9];
  const float* query_init_w = (const float*)d_in[10];
  const float* query_k_w    = (const float*)d_in[11];
  const float* query_k_b    = (const float*)d_in[12];
  const float* query_v_w    = (const float*)d_in[13];
  const float* query_v_b    = (const float*)d_in[14];
  float* out = (float*)d_out;

  char* ws = (char*)d_ws;
  float*    ze   = (float*)   (ws + 0);         // 1,048,576 B  (B,L,E) f32
  float*    ye   = (float*)   (ws + 1048576);   //   262,144 B  (B,LY,E) f32
  float*    fs32 = (float*)   (ws + 1310720);   // 1,048,576 B  (B,L,E) f32
  uint16_t* fsb  = (uint16_t*)(ws + 2359296);   //   524,288 B  (B,L,E) bf16
  uint16_t* embb = (uint16_t*)(ws + 2883584);   // 8,192,000 B  (V,E) bf16
  float*    lse  = (float*)   (ws + 11075584);  //     8,192 B  (B*L) f32

  k_cvt_bf16<<<(V_ * E_) / (256 * 4), 256, 0, stream>>>(embed_w, embb);
  k_gather_norm<<<B_ * L_ + B_ * LY_, E_, 0, stream>>>(embed_w, z, y, ze, ye);
  k_scan<<<B_, 256, SCAN_LDS_BYTES, stream>>>(
      ze, ye, init_state_w, updater_w, updater_b, updater_v_w, updater_v_b,
      query_init_w, query_k_w, query_k_b, query_v_w, query_v_b, fs32, fsb);
  k_lse<<<(B_ * L_) / 16, 256, 0, stream>>>(fsb, embb, lse);
  k_final<<<B_, 256, 0, stream>>>(fs32, embed_w, x, lse, out);
}
